// CantorAttention_46523085750349
// MI455X (gfx1250) — matrix operand order repaired and checked by reading, not compiled, not run on hardware
//
#include <hip/hip_runtime.h>
#include <math.h>

typedef _Float16 half_t;
typedef __attribute__((ext_vector_type(16))) _Float16 v16h;
typedef __attribute__((ext_vector_type(8)))  float    v8f;

#define N_TOK 2048
#define DMODEL 1024
#define NHEAD 16
#define HDIM 64
#define KNEI 128
#define SCALE 0.125f  // 1/sqrt(64)

// ---------------------------------------------------------------- converts
__global__ void ca_convert_f16(const float* __restrict__ in, half_t* __restrict__ out, int n) {
    int i = blockIdx.x * blockDim.x + threadIdx.x;
    if (i < n) out[i] = (half_t)in[i];
}

// in [R][C] f32  ->  out [C][R] f16   (K-contiguous transposed weights)
__global__ void ca_transpose_f16(const float* __restrict__ in, half_t* __restrict__ out, int R, int C) {
    __shared__ float tile[32][33];
    int bx = blockIdx.x * 32, by = blockIdx.y * 32;
    int tx = threadIdx.x, ty = threadIdx.y;           // 32 x 8
    for (int yy = ty; yy < 32; yy += 8)
        tile[yy][tx] = in[(size_t)(by + yy) * C + (bx + tx)];
    __syncthreads();
    for (int yy = ty; yy < 32; yy += 8)
        out[(size_t)(bx + yy) * R + (by + tx)] = (half_t)tile[tx][yy];
}

// ---------------------------------------------------------------- sort + routes
// One block, 1024 threads: bitonic sort of 2048 (pos, idx) pairs in LDS.
__global__ void ca_sort(const float* __restrict__ pos, float* __restrict__ spos,
                        int* __restrict__ sidx, int* __restrict__ rank) {
    __shared__ float key[N_TOK];
    __shared__ int   val[N_TOK];
    int t = threadIdx.x;
    for (int i = t; i < N_TOK; i += 1024) { key[i] = pos[i]; val[i] = i; }
    __syncthreads();
    for (unsigned k = 2; k <= N_TOK; k <<= 1) {
        for (unsigned j = k >> 1; j > 0; j >>= 1) {
            unsigned low  = ((unsigned)t / j) * (j << 1) + ((unsigned)t % j);
            unsigned high = low + j;
            bool asc = ((low & k) == 0);
            float ka = key[low], kb = key[high];
            int   va = val[low], vb = val[high];
            bool sw = asc ? (ka > kb) : (ka < kb);
            if (sw) { key[low] = kb; key[high] = ka; val[low] = vb; val[high] = va; }
            __syncthreads();
        }
    }
    for (int i = t; i < N_TOK; i += 1024) {
        spos[i] = key[i];
        sidx[i] = val[i];
        rank[val[i]] = i;
    }
}

// Per query: 128 nearest by |ci - cj| == contiguous window in sorted order.
__global__ void ca_routes(const float* __restrict__ pos, const float* __restrict__ spos,
                          const int* __restrict__ sidx, const int* __restrict__ rank,
                          int* __restrict__ routes) {
    int i = blockIdx.x * blockDim.x + threadIdx.x;
    if (i >= N_TOK) return;
    float ci = pos[i];
    int lo = rank[i], hi = lo;
    for (int s = 1; s < KNEI; ++s) {
        bool left;
        if (lo == 0) left = false;
        else if (hi == N_TOK - 1) left = true;
        else left = (ci - spos[lo - 1]) <= (spos[hi + 1] - ci);
        if (left) --lo; else ++hi;
    }
    for (int s = 0; s < KNEI; ++s) routes[(size_t)i * KNEI + s] = sidx[lo + s];
}

// ---------------------------------------------------------------- WMMA GEMM
// C[M,Ncol] = A[M,K] @ Bt[Ncol,K]^T + bias
// A row-major f16, Bt is B transposed (K contiguous) f16.
// Double-buffered LDS tiles filled with global_load_async_to_lds_b128 (ASYNCcnt),
// so next-tile staging overlaps the 8 WMMAs of the current tile.
// mode 0: store f16 into qkv layout [chan=col>>6][row][col&63], chan stride M*64
// mode 1: store f32 row-major [M][Ncol]
#define BM 128
#define BN 128
#define BK 32

union CaFrag { v16h v; uint4 q[2]; };

__global__ __launch_bounds__(256) void ca_gemm(const half_t* __restrict__ A,
                                               const half_t* __restrict__ Bt,
                                               const float* __restrict__ bias,
                                               void* __restrict__ Cout,
                                               int M, int Kdim, int Ncol, int mode) {
    __shared__ half_t Atile[2][BM * BK];
    __shared__ half_t Btile[2][BN * BK];
    int tid  = threadIdx.x;
    int lane = tid & 31, wave = tid >> 5;
    int waveM = wave & 3, waveN = wave >> 2;     // 4 x 2 waves -> 32 x 64 per wave
    int rowBase = blockIdx.y * BM;
    int colBase = blockIdx.x * BN;
    int hi = lane >> 4, r16 = lane & 15;

    // --- per-thread staging assignment: chunks c0 = tid, c1 = tid+256 of 512
    // chunk c covers 8 halves (16B): row = c>>2, half-offset = (c&3)*8
    int c0 = tid, c1 = tid + 256;
    int rA0 = c0 >> 2, oA0 = (c0 & 3) * 8;
    int rA1 = c1 >> 2, oA1 = (c1 & 3) * 8;
    const half_t* aP0 = A + (size_t)(rowBase + rA0) * Kdim + oA0;
    const half_t* aP1 = A + (size_t)(rowBase + rA1) * Kdim + oA1;
    const half_t* bP0 = Bt + (size_t)(colBase + rA0) * Kdim + oA0;
    const half_t* bP1 = Bt + (size_t)(colBase + rA1) * Kdim + oA1;
    // LDS byte addresses (generic shared ptr low 32 bits == LDS offset)
    uint32_t lA0[2], lA1[2], lB0[2], lB1[2];
    for (int b = 0; b < 2; ++b) {
        lA0[b] = (uint32_t)(uintptr_t)&Atile[b][c0 * 8];
        lA1[b] = (uint32_t)(uintptr_t)&Atile[b][c1 * 8];
        lB0[b] = (uint32_t)(uintptr_t)&Btile[b][c0 * 8];
        lB1[b] = (uint32_t)(uintptr_t)&Btile[b][c1 * 8];
    }

    v8f zero = {0.f, 0.f, 0.f, 0.f, 0.f, 0.f, 0.f, 0.f};
    v8f acc[2][4];
    for (int m = 0; m < 2; ++m)
        for (int n = 0; n < 4; ++n) acc[m][n] = zero;

    auto issueTile = [&](int b, int kb) {
        asm volatile("global_load_async_to_lds_b128 %0, %1, off"
                     :: "v"(lA0[b]), "v"(aP0 + kb) : "memory");
        asm volatile("global_load_async_to_lds_b128 %0, %1, off"
                     :: "v"(lA1[b]), "v"(aP1 + kb) : "memory");
        asm volatile("global_load_async_to_lds_b128 %0, %1, off"
                     :: "v"(lB0[b]), "v"(bP0 + kb) : "memory");
        asm volatile("global_load_async_to_lds_b128 %0, %1, off"
                     :: "v"(lB1[b]), "v"(bP1 + kb) : "memory");
    };

    int T = Kdim / BK;
    issueTile(0, 0);
    for (int it = 0; it < T; ++it) {
        // my async loads for buffer (it&1) complete:
        asm volatile("s_wait_asynccnt 0x0" ::: "memory");
        // everyone's loads complete + everyone's reads of the other buffer done
        // (__syncthreads emits s_wait_dscnt 0 before the barrier):
        __syncthreads();
        if (it + 1 < T) issueTile((it + 1) & 1, (it + 1) * BK);

        int b = it & 1;
        CaFrag a[2], bf[4];
        // A frag (16x32 f16): lanes 0-15 M=0..15 K{0..7,16..23}; lanes 16-31 K{8..15,24..31}
        for (int m = 0; m < 2; ++m) {
            const uint4* p = (const uint4*)(Atile[b] + (waveM * 32 + m * 16 + r16) * BK);
            a[m].q[0] = p[hi];
            a[m].q[1] = p[2 + hi];
        }
        // B frag (32x16 f16): lanes 0-15 K{0..7,16..23}; lanes 16-31 K{8..15,24..31}, N = lane&15
        for (int n = 0; n < 4; ++n) {
            const uint4* p = (const uint4*)(Btile[b] + (waveN * 64 + n * 16 + r16) * BK);
            bf[n].q[0] = p[hi];
            bf[n].q[1] = p[2 + hi];
        }
        for (int m = 0; m < 2; ++m)
            for (int n = 0; n < 4; ++n)
                acc[m][n] = __builtin_amdgcn_wmma_f32_16x16x32_f16(
                    false, a[m].v, false, bf[n].v, (short)0, acc[m][n], false, false);
    }

    // epilogue: C frag 16x16 f32 -> lanes 0-15 N=lane M=r; lanes 16-31 N=lane-16 M=8+r
    for (int m = 0; m < 2; ++m) {
        for (int n = 0; n < 4; ++n) {
            int gRow0 = rowBase + waveM * 32 + m * 16 + hi * 8;
            int col   = colBase + waveN * 64 + n * 16 + r16;
            float bv  = bias ? bias[col] : 0.f;
            for (int r = 0; r < 8; ++r) {
                float valf = acc[m][n][r] + bv;
                int row = gRow0 + r;
                if (mode == 0) {
                    int chan = col >> 6, hd = col & 63;
                    ((half_t*)Cout)[((size_t)chan * M + row) * 64 + hd] = (half_t)valf;
                } else {
                    ((float*)Cout)[(size_t)row * Ncol + col] = valf;
                }
            }
        }
    }
}

// ---------------------------------------------------------------- attention
// qkv layout f16 [48][N][64]: q=chan h, k=chan 16+h, v=chan 32+h.
// grid (N, H), block 128: thread t owns neighbor t.
__global__ __launch_bounds__(128) void ca_attn(const half_t* __restrict__ qkv,
                                               const int* __restrict__ routes,
                                               half_t* __restrict__ attnout) {
    __shared__ float  sq[HDIM];
    __shared__ float  sattn[KNEI];
    __shared__ float  red[KNEI];
    __shared__ half_t sv[KNEI * HDIM];
    __shared__ float  partial[2][HDIM];

    int i = blockIdx.x, h = blockIdx.y, t = threadIdx.x;
    const half_t* qrow = qkv + ((size_t)h * N_TOK + i) * HDIM;
    if (t < HDIM) sq[t] = (float)qrow[t];

    int route = routes[(size_t)i * KNEI + t];
    const half_t* krow = qkv + ((size_t)(NHEAD + h) * N_TOK + route) * HDIM;
    const half_t* vrow = qkv + ((size_t)(2 * NHEAD + h) * N_TOK + route) * HDIM;

    // stage my V row into LDS (64 halves = 8 x uint4)
    {
        const uint4* v4  = (const uint4*)vrow;
        uint4*       sv4 = (uint4*)(sv + t * HDIM);
        for (int c = 0; c < 8; ++c) sv4[c] = v4[c];
    }
    __syncthreads();

    float dot = 0.f;
    for (int d = 0; d < HDIM; ++d) dot += sq[d] * (float)krow[d];
    float score = dot * SCALE;

    // block max
    red[t] = score; __syncthreads();
    for (int s = 64; s > 0; s >>= 1) {
        if (t < s) red[t] = fmaxf(red[t], red[t + s]);
        __syncthreads();
    }
    float mx = red[0];
    __syncthreads();

    float e = __expf(score - mx);
    sattn[t] = e; red[t] = e; __syncthreads();
    for (int s = 64; s > 0; s >>= 1) {
        if (t < s) red[t] += red[t + s];
        __syncthreads();
    }
    float inv = 1.f / red[0];

    // out[d] = sum_j attn_j * v_j[d]; split neighbors across two half-blocks
    int d = t & (HDIM - 1), hsel = t >> 6;
    float o = 0.f;
    for (int j = 0; j < 64; ++j) {
        int jj = hsel * 64 + j;
        o += sattn[jj] * (float)sv[jj * HDIM + d];
    }
    partial[hsel][d] = o * inv;
    __syncthreads();
    if (t < HDIM)
        attnout[(size_t)i * DMODEL + h * HDIM + t] = (half_t)(partial[0][t] + partial[1][t]);
}

// ---------------------------------------------------------------- launch
extern "C" void kernel_launch(void* const* d_in, const int* in_sizes, int n_in,
                              void* d_out, int out_size, void* d_ws, size_t ws_size,
                              hipStream_t stream) {
    const float* x    = (const float*)d_in[0];
    const float* cant = (const float*)d_in[1];
    const float* Wqkv = (const float*)d_in[2];
    const float* bqkv = (const float*)d_in[3];
    const float* Wout = (const float*)d_in[4];
    const float* bout = (const float*)d_in[5];
    float* out = (float*)d_out;

    char* ws = (char*)d_ws;
    size_t off = 0;
    auto alloc = [&](size_t bytes) { void* p = ws + off; off = (off + bytes + 255) & ~(size_t)255; return p; };

    half_t* xh      = (half_t*)alloc((size_t)N_TOK * DMODEL * 2);          // 4 MB
    half_t* Wqkvt   = (half_t*)alloc((size_t)3 * DMODEL * DMODEL * 2);     // 6 MB
    half_t* Woutt   = (half_t*)alloc((size_t)DMODEL * DMODEL * 2);         // 2 MB
    half_t* qkv     = (half_t*)alloc((size_t)48 * N_TOK * HDIM * 2);       // 12 MB
    half_t* attnout = (half_t*)alloc((size_t)N_TOK * DMODEL * 2);          // 4 MB
    float*  spos    = (float*)alloc(N_TOK * 4);
    int*    sidx    = (int*)alloc(N_TOK * 4);
    int*    rank    = (int*)alloc(N_TOK * 4);
    int*    routes  = (int*)alloc((size_t)N_TOK * KNEI * 4);               // 1 MB

    // 1) convert x to f16
    ca_convert_f16<<<(N_TOK * DMODEL + 255) / 256, 256, 0, stream>>>(x, xh, N_TOK * DMODEL);
    // 2) transpose weights to [Ncol][K] f16
    ca_transpose_f16<<<dim3(3 * DMODEL / 32, DMODEL / 32), dim3(32, 8), 0, stream>>>(Wqkv, Wqkvt, DMODEL, 3 * DMODEL);
    ca_transpose_f16<<<dim3(DMODEL / 32, DMODEL / 32), dim3(32, 8), 0, stream>>>(Wout, Woutt, DMODEL, DMODEL);
    // 3) sort positions, build routes (nearest-128 window in sorted order)
    ca_sort<<<1, 1024, 0, stream>>>(cant, spos, sidx, rank);
    ca_routes<<<N_TOK / 128, 128, 0, stream>>>(cant, spos, sidx, rank, routes);
    // 4) QKV GEMM (WMMA, async-LDS double buffered) -> qkv f16 [48][N][64]
    ca_gemm<<<dim3(3 * DMODEL / BN, N_TOK / BM), 256, 0, stream>>>(xh, Wqkvt, bqkv, qkv,
                                                                   N_TOK, DMODEL, 3 * DMODEL, 0);
    // 5) gathered attention -> attnout f16 [N][D]
    ca_attn<<<dim3(N_TOK, NHEAD), 128, 0, stream>>>(qkv, routes, attnout);
    // 6) output GEMM (WMMA) -> d_out f32 [N][D]
    ca_gemm<<<dim3(DMODEL / BN, N_TOK / BM), 256, 0, stream>>>(attnout, Woutt, bout, out,
                                                               N_TOK, DMODEL, DMODEL, 1);
}
